// Combiner_10196252361008
// MI455X (gfx1250) — matrix operand order repaired and checked by reading, not compiled, not run on hardware
//
#include <hip/hip_runtime.h>
#include <hip/hip_bf16.h>

#define BATCH 128
#define NDIM  2048
#define CDIM  512
#define JDIM  64
#define KCLS  200

typedef __attribute__((ext_vector_type(16))) __bf16 v16bf;
typedef __attribute__((ext_vector_type(8)))  float  v8f;
typedef __attribute__((ext_vector_type(2)))  float  v2f;
typedef __attribute__((ext_vector_type(2)))  __bf16 v2bf;

union BF16x16 { v16bf v; unsigned u[8]; };

// single v_cvt_pk_bf16_f32: two f32 -> packed bf16 pair in one dword
__device__ __forceinline__ unsigned cvt2(float lo, float hi) {
    v2f f = {lo, hi};
    v2bf b = __builtin_convertvector(f, v2bf);
    union { v2bf b; unsigned u; } p; p.b = b;
    return p.u;
}

__device__ __forceinline__ v8f v8f_zero() {
    v8f z = {0.f,0.f,0.f,0.f,0.f,0.f,0.f,0.f};
    return z;
}

// ---------------------------------------------------------------------------
// K1: Y[b,j,c] = sum_n w_pool0[j,n] * x[b,n,c]  + b_pool0[j]
// Per block: batch b, 128-wide c tile, all 64 j. WMMA bf16, K=n loop,
// B (x tile) staged through LDS in fragment layout, double buffered and
// software pipelined (next tile's global loads issued before the barrier).
// ---------------------------------------------------------------------------
__global__ __launch_bounds__(256) void k1_pool0(
    const float* __restrict__ x, const float* __restrict__ w_pool0,
    const float* __restrict__ b_pool0, float* __restrict__ Y)
{
    __shared__ __align__(32) unsigned ldsB[2][8][32][8];   // 16KB: [buf][csub][lane][kpair]

    const int b     = blockIdx.y;
    const int c0    = blockIdx.x * 128;
    const int t     = threadIdx.x;
    const int lane  = t & 31;
    const int wv    = t >> 5;
    const int m     = (wv & 3) * 16;     // j tile base
    const int chalf = wv >> 2;           // 0/1 -> 64 c columns
    const int khalf = lane >> 4;
    const int l16   = lane & 15;

    v8f acc[4];
    #pragma unroll
    for (int i = 0; i < 4; ++i) acc[i] = v8f_zero();

    const float* xb = x + (size_t)b * NDIM * CDIM;

    // staging coordinates (2 items per thread, 512 items = 32k x 128c tile)
    int cg[2], kp[2];
    #pragma unroll
    for (int it = 0; it < 2; ++it) {
        int w = t + 256 * it;
        cg[it] = w & 31;      // c group of 4
        kp[it] = w >> 5;      // k pair index
    }

    // prologue: global loads for k0 = 0
    float4 f0[2], f1[2];
    #pragma unroll
    for (int it = 0; it < 2; ++it) {
        const float* p0 = xb + (size_t)(kp[it] * 2) * CDIM + c0 + cg[it] * 4;
        f0[it] = *(const float4*)p0;
        f1[it] = *(const float4*)(p0 + CDIM);
    }

    for (int k0 = 0; k0 < NDIM; k0 += 32) {
        const int buf = (k0 >> 5) & 1;

        // ---- stage current tile: 4x v_cvt_pk_bf16_f32 + stores per item ----
        #pragma unroll
        for (int it = 0; it < 2; ++it) {
            int cl = cg[it] * 4, kl = kp[it] * 2;
            const int lrow = (kl >> 4) << 4;
            const int kidx = (kl & 15) >> 1;
            unsigned p0v = cvt2(f0[it].x, f1[it].x);
            unsigned p1v = cvt2(f0[it].y, f1[it].y);
            unsigned p2v = cvt2(f0[it].z, f1[it].z);
            unsigned p3v = cvt2(f0[it].w, f1[it].w);
            ldsB[buf][(cl + 0) >> 4][((cl + 0) & 15) + lrow][kidx] = p0v;
            ldsB[buf][(cl + 1) >> 4][((cl + 1) & 15) + lrow][kidx] = p1v;
            ldsB[buf][(cl + 2) >> 4][((cl + 2) & 15) + lrow][kidx] = p2v;
            ldsB[buf][(cl + 3) >> 4][((cl + 3) & 15) + lrow][kidx] = p3v;
        }

        // ---- software pipeline: issue next tile's global loads now ----
        if (k0 + 32 < NDIM) {
            #pragma unroll
            for (int it = 0; it < 2; ++it) {
                const float* p0 = xb + (size_t)(k0 + 32 + kp[it] * 2) * CDIM + c0 + cg[it] * 4;
                f0[it] = *(const float4*)p0;
                f1[it] = *(const float4*)(p0 + CDIM);
            }
        }
        // prefetch two K-steps ahead into caches (global_prefetch_b8)
        if (k0 + 64 < NDIM) {
            __builtin_prefetch(xb + (size_t)(k0 + 64 + kp[0] * 2) * CDIM + c0 + cg[0] * 4, 0, 0);
        }
        __syncthreads();

        // ---- A fragment: w_pool0 rows (contiguous K), packed hw cvt ----
        BF16x16 afr;
        {
            const float* ap = w_pool0 + (size_t)(m + l16) * NDIM + k0 + khalf * 16;
            float4 a0 = *(const float4*)(ap);
            float4 a1 = *(const float4*)(ap + 4);
            float4 a2 = *(const float4*)(ap + 8);
            float4 a3 = *(const float4*)(ap + 12);
            afr.u[0] = cvt2(a0.x, a0.y); afr.u[1] = cvt2(a0.z, a0.w);
            afr.u[2] = cvt2(a1.x, a1.y); afr.u[3] = cvt2(a1.z, a1.w);
            afr.u[4] = cvt2(a2.x, a2.y); afr.u[5] = cvt2(a2.z, a2.w);
            afr.u[6] = cvt2(a3.x, a3.y); afr.u[7] = cvt2(a3.z, a3.w);
        }

        // ---- preload all 4 B fragments, then back-to-back WMMA ----
        v16bf bfr[4];
        #pragma unroll
        for (int i = 0; i < 4; ++i)
            bfr[i] = *(const v16bf*)&ldsB[buf][chalf * 4 + i][lane][0];
        #pragma unroll
        for (int i = 0; i < 4; ++i)
            acc[i] = __builtin_amdgcn_wmma_f32_16x16x32_bf16(
                false, afr.v, false, bfr[i], (short)0, acc[i], false, false);
    }

    float* yb = Y + (size_t)b * JDIM * CDIM;
    #pragma unroll
    for (int i = 0; i < 4; ++i) {
        int c = c0 + chalf * 64 + i * 16 + l16;
        #pragma unroll
        for (int r = 0; r < 8; ++r) {
            int j = m + r + (khalf << 3);
            yb[(size_t)j * CDIM + c] = acc[i][r] + b_pool0[j];
        }
    }
}

// ---------------------------------------------------------------------------
// K2a: column means of w_q / w_k over the 128 Q rows, plus bias means
// ---------------------------------------------------------------------------
__global__ void k2a_wbar(const float* __restrict__ w_q, const float* __restrict__ b_q,
                         const float* __restrict__ w_k, const float* __restrict__ b_k,
                         float* __restrict__ wbarq, float* __restrict__ wbark,
                         float* __restrict__ bmeans)
{
    int c = threadIdx.x;   // 512 threads
    float sq = 0.f, sk = 0.f;
    for (int q = 0; q < 128; ++q) { sq += w_q[q * CDIM + c]; sk += w_k[q * CDIM + c]; }
    wbarq[c] = sq * (1.f / 128.f);
    wbark[c] = sk * (1.f / 128.f);
    if (c == 0) {
        float s1 = 0.f, s2 = 0.f;
        for (int q = 0; q < 128; ++q) { s1 += b_q[q]; s2 += b_k[q]; }
        bmeans[0] = s1 * (1.f / 128.f);
        bmeans[1] = s2 * (1.f / 128.f);
    }
}

// ---------------------------------------------------------------------------
// K2b: q1[b,j] = sum_c wbarq[c]*Y[b,j,c] + mean(b_q)   (same for k1)
// ---------------------------------------------------------------------------
__global__ __launch_bounds__(256) void k2b_qk(
    const float* __restrict__ Y, const float* __restrict__ wbarq,
    const float* __restrict__ wbark, const float* __restrict__ bmeans,
    float* __restrict__ q1, float* __restrict__ k1)
{
    __shared__ float wq[CDIM], wk[CDIM], rq[256], rk[256];
    const int b = blockIdx.x, t = threadIdx.x;
    wq[t] = wbarq[t]; wq[t + 256] = wbarq[t + 256];
    wk[t] = wbark[t]; wk[t + 256] = wbark[t + 256];
    __syncthreads();
    const int j = t >> 2, p = t & 3;
    const float* yr = Y + ((size_t)b * JDIM + j) * CDIM + p * 128;
    float aq = 0.f, ak = 0.f;
    for (int c = 0; c < 128; ++c) {
        float y = yr[c];
        aq += y * wq[p * 128 + c];
        ak += y * wk[p * 128 + c];
    }
    rq[t] = aq; rk[t] = ak;
    __syncthreads();
    if (p == 0) {
        q1[b * JDIM + j] = rq[t] + rq[t+1] + rq[t+2] + rq[t+3] + bmeans[0];
        k1[b * JDIM + j] = rk[t] + rk[t+1] + rk[t+2] + rk[t+3] + bmeans[1];
    }
}

// ---------------------------------------------------------------------------
// K3: Z[b,j,o] = sum_c Y[b,j,c] * w_conv1[o,c] + b_conv1[o]
// Both operands K-contiguous -> fragments loaded directly from global.
// ---------------------------------------------------------------------------
__global__ __launch_bounds__(256) void k3_conv1(
    const float* __restrict__ Y, const float* __restrict__ w_conv1,
    const float* __restrict__ b_conv1, float* __restrict__ Z)
{
    const int b     = blockIdx.y;
    const int o0    = blockIdx.x * 128;
    const int t     = threadIdx.x;
    const int lane  = t & 31;
    const int wv    = t >> 5;
    const int m     = (wv & 3) * 16;
    const int ohalf = wv >> 2;
    const int khalf = lane >> 4;
    const int l16   = lane & 15;

    v8f acc[4];
    #pragma unroll
    for (int i = 0; i < 4; ++i) acc[i] = v8f_zero();

    const float* yb = Y + (size_t)b * JDIM * CDIM;

    for (int k0 = 0; k0 < CDIM; k0 += 32) {
        BF16x16 afr;
        {
            const float* ap = yb + (size_t)(m + l16) * CDIM + k0 + khalf * 16;
            float4 a0 = *(const float4*)(ap);
            float4 a1 = *(const float4*)(ap + 4);
            float4 a2 = *(const float4*)(ap + 8);
            float4 a3 = *(const float4*)(ap + 12);
            afr.u[0] = cvt2(a0.x, a0.y); afr.u[1] = cvt2(a0.z, a0.w);
            afr.u[2] = cvt2(a1.x, a1.y); afr.u[3] = cvt2(a1.z, a1.w);
            afr.u[4] = cvt2(a2.x, a2.y); afr.u[5] = cvt2(a2.z, a2.w);
            afr.u[6] = cvt2(a3.x, a3.y); afr.u[7] = cvt2(a3.z, a3.w);
        }
        BF16x16 bfr[4];
        #pragma unroll
        for (int i = 0; i < 4; ++i) {
            int o = o0 + ohalf * 64 + i * 16 + l16;
            const float* bp = w_conv1 + (size_t)o * CDIM + k0 + khalf * 16;
            float4 b0 = *(const float4*)(bp);
            float4 b1 = *(const float4*)(bp + 4);
            float4 b2 = *(const float4*)(bp + 8);
            float4 b3 = *(const float4*)(bp + 12);
            bfr[i].u[0] = cvt2(b0.x, b0.y); bfr[i].u[1] = cvt2(b0.z, b0.w);
            bfr[i].u[2] = cvt2(b1.x, b1.y); bfr[i].u[3] = cvt2(b1.z, b1.w);
            bfr[i].u[4] = cvt2(b2.x, b2.y); bfr[i].u[5] = cvt2(b2.z, b2.w);
            bfr[i].u[6] = cvt2(b3.x, b3.y); bfr[i].u[7] = cvt2(b3.z, b3.w);
        }
        #pragma unroll
        for (int i = 0; i < 4; ++i)
            acc[i] = __builtin_amdgcn_wmma_f32_16x16x32_bf16(
                false, afr.v, false, bfr[i].v, (short)0, acc[i], false, false);
    }

    float* zb = Z + (size_t)b * JDIM * CDIM;
    #pragma unroll
    for (int i = 0; i < 4; ++i) {
        int o = o0 + ohalf * 64 + i * 16 + l16;
        float bias = b_conv1[o];
        #pragma unroll
        for (int r = 0; r < 8; ++r) {
            int j = m + r + (khalf << 3);
            zb[(size_t)j * CDIM + o] = acc[i][r] + bias;
        }
    }
}

// ---------------------------------------------------------------------------
// K4: Z3[b,j,c] = sum_jj A1[b,jj,j] * Z[b,jj,c]
//     A1[jj,j] = adj1[jj,j] + tanh(q1[jj]-k1[j])*alpha  (A^T fragments in LDS)
//     Fused BN statistic accumulation (per-channel sum / sumsq over b,j).
// ---------------------------------------------------------------------------
__global__ __launch_bounds__(256) void k4_adj(
    const float* __restrict__ Z, const float* __restrict__ adj1,
    const float* __restrict__ q1, const float* __restrict__ k1,
    const float* __restrict__ alpha1, float* __restrict__ Z3,
    float* __restrict__ gsum, float* __restrict__ gsum2)
{
    __shared__ __align__(32) unsigned a1f[2][4][32][8];  // [kstep][mtile][lane][kpair]
    __shared__ float q1s[JDIM], k1s[JDIM];
    __shared__ float ssum[128], ssum2[128];

    const int b     = blockIdx.y;
    const int c0    = blockIdx.x * 128;
    const int t     = threadIdx.x;
    const int lane  = t & 31;
    const int wv    = t >> 5;
    const int m     = (wv & 3) * 16;
    const int chalf = wv >> 2;
    const int khalf = lane >> 4;
    const int l16   = lane & 15;

    if (t < 64) { q1s[t] = q1[b * JDIM + t]; k1s[t] = k1[b * JDIM + t]; }
    if (t < 128) { ssum[t] = 0.f; ssum2[t] = 0.f; }
    const float alpha = alpha1[0];
    __syncthreads();

    // Build A1^T bf16 fragments in LDS: row j, K = jj
    #pragma unroll
    for (int it = 0; it < 8; ++it) {
        int w  = t + 256 * it;
        int j  = w & 63;
        int pr = w >> 6;               // jj pair index 0..31
        int jj0 = pr * 2;
        int ks = jj0 >> 5;
        int jr = jj0 & 31;
        float v0 = adj1[(ks * 32 + jr)     * JDIM + j] + tanhf(q1s[ks * 32 + jr]     - k1s[j]) * alpha;
        float v1 = adj1[(ks * 32 + jr + 1) * JDIM + j] + tanhf(q1s[ks * 32 + jr + 1] - k1s[j]) * alpha;
        int lane2 = (j & 15) + 16 * (jr >> 4);
        int kk    = jr & 15;
        a1f[ks][j >> 4][lane2][kk >> 1] = cvt2(v0, v1);
    }
    __syncthreads();

    v8f acc[4];
    #pragma unroll
    for (int i = 0; i < 4; ++i) acc[i] = v8f_zero();

    const float* zb = Z + (size_t)b * JDIM * CDIM;
    #pragma unroll
    for (int ks = 0; ks < 2; ++ks) {
        v16bf afr = *(const v16bf*)&a1f[ks][wv & 3][lane][0];
        #pragma unroll
        for (int i = 0; i < 4; ++i) {
            int c = c0 + chalf * 64 + i * 16 + l16;
            const float* zp = zb + (size_t)(ks * 32 + khalf * 16) * CDIM + c;
            BF16x16 bfr;
            #pragma unroll
            for (int kk2 = 0; kk2 < 8; ++kk2) {
                float z0 = zp[(size_t)(2 * kk2)     * CDIM];
                float z1 = zp[(size_t)(2 * kk2 + 1) * CDIM];
                bfr.u[kk2] = cvt2(z0, z1);
            }
            acc[i] = __builtin_amdgcn_wmma_f32_16x16x32_bf16(
                false, afr, false, bfr.v, (short)0, acc[i], false, false);
        }
    }

    float* z3b = Z3 + (size_t)b * JDIM * CDIM;
    #pragma unroll
    for (int i = 0; i < 4; ++i) {
        int cl = chalf * 64 + i * 16 + l16;
        int c  = c0 + cl;
        float s = 0.f, s2 = 0.f;
        #pragma unroll
        for (int r = 0; r < 8; ++r) {
            int j = m + r + (khalf << 3);
            float v = acc[i][r];
            z3b[(size_t)j * CDIM + c] = v;
            s += v; s2 += v * v;
        }
        atomicAdd(&ssum[cl], s);
        atomicAdd(&ssum2[cl], s2);
    }
    __syncthreads();
    if (t < 128) {
        atomicAdd(&gsum[c0 + t],  ssum[t]);
        atomicAdd(&gsum2[c0 + t], ssum2[t]);
    }
}

// ---------------------------------------------------------------------------
// K5: finalize batchnorm statistics -> per-channel scale/shift
// ---------------------------------------------------------------------------
__global__ void k5_bn(const float* __restrict__ gsum, const float* __restrict__ gsum2,
                      const float* __restrict__ gamma, const float* __restrict__ beta,
                      float* __restrict__ scaleA, float* __restrict__ shiftA)
{
    int c = threadIdx.x;   // 512 threads
    const float invn = 1.0f / (float)(BATCH * JDIM);
    float mean = gsum[c] * invn;
    float var  = gsum2[c] * invn - mean * mean;
    float sc   = gamma[c] * rsqrtf(var + 1e-5f);
    scaleA[c] = sc;
    shiftA[c] = beta[c] - mean * sc;
}

// ---------------------------------------------------------------------------
// K6: fused BN affine + pool over J + classifier head
// ---------------------------------------------------------------------------
__global__ __launch_bounds__(256) void k6_head(
    const float* __restrict__ Z3, const float* __restrict__ scaleA,
    const float* __restrict__ shiftA, const float* __restrict__ w_pool1,
    const float* __restrict__ b_pool1, const float* __restrict__ w_cls,
    const float* __restrict__ b_cls, float* __restrict__ out)
{
    __shared__ float wp[JDIM];
    __shared__ float pooled[CDIM];
    const int b = blockIdx.x, t = threadIdx.x;
    if (t < JDIM) wp[t] = w_pool1[t];
    __syncthreads();
    float swp = 0.f;
    #pragma unroll
    for (int j = 0; j < JDIM; ++j) swp += wp[j];

    const float* z3b = Z3 + (size_t)b * JDIM * CDIM;
    #pragma unroll
    for (int i = 0; i < 2; ++i) {
        int c = t + 256 * i;
        float acc = 0.f;
        for (int j = 0; j < JDIM; ++j) acc += z3b[(size_t)j * CDIM + c] * wp[j];
        pooled[c] = scaleA[c] * acc + shiftA[c] * swp + b_pool1[0];
    }
    __syncthreads();
    if (t < KCLS) {
        float acc = b_cls[t];
        const float* wr = w_cls + (size_t)t * CDIM;
        for (int c = 0; c < CDIM; ++c) acc += pooled[c] * wr[c];
        out[b * KCLS + t] = acc;
    }
}

// ---------------------------------------------------------------------------
extern "C" void kernel_launch(void* const* d_in, const int* in_sizes, int n_in,
                              void* d_out, int out_size, void* d_ws, size_t ws_size,
                              hipStream_t stream) {
    const float* x       = (const float*)d_in[0];
    const float* w_pool0 = (const float*)d_in[1];
    const float* b_pool0 = (const float*)d_in[2];
    const float* adj1    = (const float*)d_in[3];
    const float* w_conv1 = (const float*)d_in[4];
    const float* b_conv1 = (const float*)d_in[5];
    const float* w_q     = (const float*)d_in[6];
    const float* b_q     = (const float*)d_in[7];
    const float* w_k     = (const float*)d_in[8];
    const float* b_k     = (const float*)d_in[9];
    const float* alpha1  = (const float*)d_in[10];
    const float* gamma   = (const float*)d_in[11];
    const float* beta    = (const float*)d_in[12];
    const float* w_pool1 = (const float*)d_in[13];
    const float* b_pool1 = (const float*)d_in[14];
    const float* w_cls   = (const float*)d_in[15];
    const float* b_cls   = (const float*)d_in[16];
    float* out = (float*)d_out;

    char* ws = (char*)d_ws;
    float* Y  = (float*)ws;                            // 16MB [B,J,C]; reused as Z3
    float* Z  = (float*)(ws + ((size_t)16 << 20));     // 16MB [B,J,C]
    float* sm = (float*)(ws + ((size_t)32 << 20));
    float* wbarq  = sm;                 // 512
    float* wbark  = sm + 512;           // 512
    float* q1     = sm + 1024;          // 8192
    float* k1     = sm + 1024 + 8192;   // 8192
    float* gsum   = sm + 1024 + 16384;  // 512
    float* gsum2  = gsum + 512;         // 512 (contiguous with gsum)
    float* scaleA = gsum2 + 512;        // 512
    float* shiftA = scaleA + 512;       // 512
    float* bmeans = shiftA + 512;       // 2

    k1_pool0<<<dim3(4, BATCH), 256, 0, stream>>>(x, w_pool0, b_pool0, Y);
    k2a_wbar<<<1, 512, 0, stream>>>(w_q, b_q, w_k, b_k, wbarq, wbark, bmeans);
    k2b_qk<<<BATCH, 256, 0, stream>>>(Y, wbarq, wbark, bmeans, q1, k1);
    k3_conv1<<<dim3(4, BATCH), 256, 0, stream>>>(Y, w_conv1, b_conv1, Z);
    hipMemsetAsync(gsum, 0, 2 * CDIM * sizeof(float), stream);
    k4_adj<<<dim3(4, BATCH), 256, 0, stream>>>(Z, adj1, q1, k1, alpha1, /*Z3=*/Y, gsum, gsum2);
    k5_bn<<<1, 512, 0, stream>>>(gsum, gsum2, gamma, beta, scaleA, shiftA);
    k6_head<<<BATCH, 256, 0, stream>>>(/*Z3=*/Y, scaleA, shiftA, w_pool1, b_pool1, w_cls, b_cls, out);
}
